// AugmentedLstm_61349312856223
// MI455X (gfx1250) — matrix operand order repaired and checked by reading, not compiled, not run on hardware
//
#include <hip/hip_runtime.h>
#include <stdint.h>

// ---- CDNA5 WMMA types -------------------------------------------------------
typedef __attribute__((ext_vector_type(16))) __bf16       bf16x16;
typedef __attribute__((ext_vector_type(8)))  float        floatx8;
typedef __attribute__((ext_vector_type(4))) unsigned int uint4v;

#define BB   128
#define TT   512
#define HH   512
#define N6   3072
#define NBLK 32          // persistent workgroups == H/16 column slices
#define WROW 104         // LDS row stride in shorts: 96 data + 8 pad = 208 B (odd multiple of 16 B)

// fp32 -> bf16 RNE (scalar, one-time W staging only)
__device__ __forceinline__ unsigned short f2bf(float f) {
    union { float f; unsigned u; } v; v.f = f;
    return (unsigned short)((v.u + 0x7FFFu + ((v.u >> 16) & 1u)) >> 16);
}
// pack two fp32 -> dword of two bf16, round-half-up: 2x v_add + 1x v_perm_b32
__device__ __forceinline__ unsigned f2bf_pk(float lo, float hi) {
    union { float f; unsigned u; } a, b; a.f = lo; b.f = hi;
    return __builtin_amdgcn_perm(b.u + 0x8000u, a.u + 0x8000u, 0x07060302u);
}
// fast sigmoid / tanh: v_exp_f32 + single v_rcp_f32 (saturate correctly at +-inf)
__device__ __forceinline__ float sigm(float x) {
    return __builtin_amdgcn_rcpf(1.0f + __expf(-x));
}
__device__ __forceinline__ float tanh_fast(float x) {
    return 1.0f - 2.0f * __builtin_amdgcn_rcpf(__expf(2.0f * x) + 1.0f);
}

union Frag {
    unsigned d[8];
    struct { uint4v lo, hi; } q;
    bf16x16 v;
};

// ---- init: zero grid-barrier slots + the h==0 row (fresh every replay) ------
__global__ void init_ws_kernel(int* __restrict__ slots, float* __restrict__ zrow) {
    slots[threadIdx.x] = 0;          // <<<1, 512>>>
    zrow[threadIdx.x]  = 0.0f;
}

// ---- persistent fused LSTM: 32 blocks x 8 waves, all 512 timesteps ----------
__launch_bounds__(256, 1)
__global__ void lstm_persistent_kernel(const float* __restrict__ x,       // [128,512,512]
                                       const float* __restrict__ W,       // [512,3072] fp32
                                       const float* __restrict__ bias,    // [3072]
                                       const int*  __restrict__ lengths,  // [128]
                                       float* __restrict__ out,           // [128,512,512] = h history
                                       int*   __restrict__ slots,         // [512] barrier counters
                                       const float* __restrict__ zrow) {  // [512] zeros (h at t=0)
    extern __shared__ unsigned short wl[];            // [512][WROW] bf16 W slice

    const int lane   = threadIdx.x & 31;
    const int wave   = threadIdx.x >> 5;
    const int m0     = wave * 16;                     // batch-row tile
    const int j0     = blockIdx.x * 16;               // H-column tile
    const int mrow   = lane & 15;
    const int hiHalf = lane >> 4;
    const int ksub   = hiHalf * 8;                    // ISA 16-bit A-fragment K split

    // ---- stage this block's W slice into LDS once (fp32 -> bf16) -----------
    for (int idx = threadIdx.x; idx < 512 * 96; idx += 256) {
        int k = idx / 96, col = idx - k * 96;
        int g = col >> 4, cc = col & 15;
        wl[k * WROW + col] = f2bf(W[(size_t)k * N6 + g * HH + j0 + cc]);
    }
    __syncthreads();

    // ---- per-lane invariants ------------------------------------------------
    const int   j   = j0 + mrow;                      // this lane's output column
    const float bi2 = 2.0f * bias[0 * HH + j];
    const float bf2 = 2.0f * bias[1 * HH + j];
    const float bm2 = 2.0f * bias[2 * HH + j];
    const float bo2 = 2.0f * bias[3 * HH + j];
    const float bh2 = 2.0f * bias[4 * HH + j];
    const float b5  =        bias[5 * HH + j];
    int lenr[8];
#pragma unroll
    for (int r = 0; r < 8; ++r) lenr[r] = lengths[m0 + r + 8 * hiHalf];

    floatx8 creg = {};                                // c-state tile lives in VGPRs all run

    const size_t arow = (size_t)(m0 + mrow) * (TT * HH);   // this lane's A row in x/out

    for (int t = 0; t < TT; ++t) {
        floatx8 acc[6] = {};                          // i, f, m, o, hw, px5
        const float* xrow = x + arow + (size_t)t * HH;
        // t==0: read the zero row instead of branching inside the K loop
        const float* hrow = (t > 0) ? (out + arow + (size_t)(t - 1) * HH) : zrow;

#pragma unroll 2
        for (int kc = 0; kc < HH; kc += 32) {
            // ---- build A fragments in-register: A_xh = bf16(x+h), A_x = bf16(x)
            Frag axh, ax, bfr;
            const float* xp = xrow + kc + ksub;
            const float* hp = hrow + kc + ksub;
            float4 xa = *(const float4*)(xp);
            float4 xb = *(const float4*)(xp + 4);
            float4 xc = *(const float4*)(xp + 16);
            float4 xd = *(const float4*)(xp + 20);
            float4 ha = *(const float4*)(hp);
            float4 hb = *(const float4*)(hp + 4);
            float4 hc = *(const float4*)(hp + 16);
            float4 hd = *(const float4*)(hp + 20);

            ax.d[0] = f2bf_pk(xa.x, xa.y);  ax.d[1] = f2bf_pk(xa.z, xa.w);
            ax.d[2] = f2bf_pk(xb.x, xb.y);  ax.d[3] = f2bf_pk(xb.z, xb.w);
            ax.d[4] = f2bf_pk(xc.x, xc.y);  ax.d[5] = f2bf_pk(xc.z, xc.w);
            ax.d[6] = f2bf_pk(xd.x, xd.y);  ax.d[7] = f2bf_pk(xd.z, xd.w);
            axh.d[0] = f2bf_pk(xa.x + ha.x, xa.y + ha.y);
            axh.d[1] = f2bf_pk(xa.z + ha.z, xa.w + ha.w);
            axh.d[2] = f2bf_pk(xb.x + hb.x, xb.y + hb.y);
            axh.d[3] = f2bf_pk(xb.z + hb.z, xb.w + hb.w);
            axh.d[4] = f2bf_pk(xc.x + hc.x, xc.y + hc.y);
            axh.d[5] = f2bf_pk(xc.z + hc.z, xc.w + hc.w);
            axh.d[6] = f2bf_pk(xd.x + hd.x, xd.y + hd.y);
            axh.d[7] = f2bf_pk(xd.z + hd.z, xd.w + hd.w);

            // ---- six 16x16x32 WMMAs, B fragments from LDS -------------------
            const unsigned short* wrow = wl + (kc + ksub) * WROW + mrow;   // lane indexes N column, K split ksub
#pragma unroll
            for (int g = 0; g < 6; ++g) {
                const unsigned short* pw = wrow + g * 16;
#pragma unroll
                for (int i = 0; i < 8; ++i) {
                    const int k0 = (i < 4) ? 2 * i : 8 + 2 * i;
                    bfr.d[i] = (unsigned)pw[k0 * WROW] | ((unsigned)pw[(k0 + 1) * WROW] << 16);
                }
                bf16x16 amat = (g == 5) ? ax.v : axh.v;   // highway bypass: x only
                acc[g] = __builtin_amdgcn_wmma_f32_16x16x32_bf16(
                    false, amat, false, bfr.v, (short)0, acc[g], false, false);
            }
        }

        // ---- fused gates / highway / mask; c stays in registers -------------
#pragma unroll
        for (int r = 0; r < 8; ++r) {
            const int row = m0 + r + 8 * hiHalf;
            float ig = sigm(acc[0][r] + bi2);
            float fg = sigm(acc[1][r] + bf2);
            float mg = tanh_fast(acc[2][r] + bm2);
            float og = sigm(acc[3][r] + bo2);
            float hw = sigm(acc[4][r] + bh2);
            float p5 = acc[5][r] + b5;

            float cn = ig * mg + fg * creg[r];
            float o1 = og * tanh_fast(cn);
            float hn = hw * o1 + (1.0f - hw) * p5;

            bool active = (t < lenr[r]);
            creg[r] = active ? cn : 0.0f;
            out[(size_t)row * (TT * HH) + (size_t)t * HH + j] = active ? hn : 0.0f;
        }

        // ---- device-scope grid barrier: publish h_t before anyone reads it --
        if (t != TT - 1) {
            // warm L2/L0 with next step's x row while we sit in the barrier
            __builtin_prefetch(xrow + HH + ksub, 0, 3);
            __builtin_prefetch(xrow + HH + 256 + ksub, 0, 3);
            __threadfence();                 // release this wave's h stores to agent scope
            __syncthreads();                 // all waves of the block have fenced
            if (threadIdx.x == 0) {
                __hip_atomic_fetch_add(&slots[t], 1, __ATOMIC_RELEASE, __HIP_MEMORY_SCOPE_AGENT);
                while (__hip_atomic_load(&slots[t], __ATOMIC_ACQUIRE, __HIP_MEMORY_SCOPE_AGENT) < NBLK)
                    __builtin_amdgcn_s_sleep(1);
            }
            __syncthreads();                 // fan the acquire out to all waves
        }
    }
}

// ---- host driver ------------------------------------------------------------
extern "C" void kernel_launch(void* const* d_in, const int* in_sizes, int n_in,
                              void* d_out, int out_size, void* d_ws, size_t ws_size,
                              hipStream_t stream) {
    const float* x       = (const float*)d_in[0];   // [128,512,512]
    const int*   lengths = (const int*)  d_in[1];   // [128]
    const float* W       = (const float*)d_in[2];   // [512,3072]
    const float* bias    = (const float*)d_in[3];   // [3072]
    float*       out     = (float*)d_out;           // [128,512,512]

    int*   slots = (int*)d_ws;                      // [512] barrier counters
    float* zrow  = (float*)((char*)d_ws + TT * sizeof(int));   // [512] zeros

    init_ws_kernel<<<1, TT, 0, stream>>>(slots, zrow);

    const size_t lds_bytes = (size_t)512 * WROW * sizeof(unsigned short);   // ~104 KB / WGP
    lstm_persistent_kernel<<<NBLK, 256, lds_bytes, stream>>>(x, W, bias, lengths, out, slots, zrow);
}
